// Sae_57372173140183
// MI455X (gfx1250) — compile-verified, matrix operand order checked by reading, not run on hardware
//
#include <hip/hip_runtime.h>

// ---------------- types ----------------
typedef __bf16 bf16_t;
typedef __attribute__((ext_vector_type(4)))  __bf16 bf16x4;
typedef __attribute__((ext_vector_type(8)))  __bf16 bf16x8;
typedef __attribute__((ext_vector_type(16))) __bf16 v16bf;
typedef __attribute__((ext_vector_type(8)))  float  v8f;
typedef __attribute__((ext_vector_type(4)))  int    v4i;

#define GLOBAL_AS __attribute__((address_space(1)))
#define LDS_AS    __attribute__((address_space(3)))

// ---- CDNA5 async global->LDS path (guarded; falls back to vgpr staging) ----
#if defined(__has_builtin)
# if __has_builtin(__builtin_amdgcn_global_load_async_to_lds_b128)
#  define HAVE_ASYNC_LDS 1
# endif
# if __has_builtin(__builtin_amdgcn_s_wait_asynccnt)
#  define HAVE_WAIT_ASYNC_BUILTIN 1
# endif
#endif
#ifndef HAVE_ASYNC_LDS
# define HAVE_ASYNC_LDS 0
#endif
#ifndef HAVE_WAIT_ASYNC_BUILTIN
# define HAVE_WAIT_ASYNC_BUILTIN 0
#endif

#if HAVE_ASYNC_LDS
#if HAVE_WAIT_ASYNC_BUILTIN
# define WAIT_ASYNC(n) __builtin_amdgcn_s_wait_asynccnt(n)
#else
# define WAIT_ASYNC(n) asm volatile("s_wait_asynccnt %0" :: "i"(n) : "memory")
#endif
typedef GLOBAL_AS v4i gv4i_t;   // int4 in global (AS1)
typedef LDS_AS    v4i lv4i_t;   // int4 in LDS (AS3)
// generic->AS pointers via integer casts: global generic == flat VA (AS1 bit-identical);
// generic LDS pointer low 32 bits are the LDS byte offset (ISA aperture rule).
static __device__ __forceinline__ void async_b128(const void* g, void* l) {
  __builtin_amdgcn_global_load_async_to_lds_b128(
      (gv4i_t*)(unsigned long long)g,
      (lv4i_t*)(unsigned)(unsigned long long)l, 0, 0);
}
#endif

// ---------------- problem constants ----------------
constexpr int N_TOKENS = 4096;
constexpr int D_IN     = 1024;
constexpr int NLAT     = 32768;
constexpr int TOPK     = 64;

// ws layout (bytes)
constexpr size_t ACTS_BYTES = (size_t)N_TOKENS * NLAT * 4;        // 512 MB f32
constexpr size_t XB_OFF     = ACTS_BYTES;                         // bf16 x - b_dec
constexpr size_t XB_BYTES   = (size_t)N_TOKENS * D_IN * 2;
constexpr size_t WB_OFF     = XB_OFF + XB_BYTES;                  // bf16 W_enc
constexpr size_t WB_BYTES   = (size_t)NLAT * D_IN * 2;
constexpr size_t CS_OFF     = WB_OFF + WB_BYTES;                  // colsum  [1024] f32
constexpr size_t CQ_OFF     = CS_OFF + D_IN * 4;                  // colsumsq[1024] f32
constexpr size_t E2_OFF     = CQ_OFF + D_IN * 4;                  // tok_e2  [4096] f32

// d_out layout (floats)
constexpr size_t OUT_SAE  = 0;
constexpr size_t OUT_TV   = (size_t)N_TOKENS * D_IN;
constexpr size_t OUT_TI   = OUT_TV + (size_t)N_TOKENS * TOPK;
constexpr size_t OUT_FVU  = OUT_TI + (size_t)N_TOKENS * TOPK;

// ---------------- 1) convert x - b_dec -> bf16 ----------------
__global__ void k_convert_x(const float* __restrict__ x, const float* __restrict__ bdec,
                            bf16_t* __restrict__ Xb) {
  int g4 = blockIdx.x * blockDim.x + threadIdx.x;       // 4 elems per thread
  float4 v = ((const float4*)x)[g4];
  int d0 = (g4 * 4) & (D_IN - 1);
  bf16x4 o;
  o[0] = (bf16_t)(v.x - bdec[d0 + 0]);
  o[1] = (bf16_t)(v.y - bdec[d0 + 1]);
  o[2] = (bf16_t)(v.z - bdec[d0 + 2]);
  o[3] = (bf16_t)(v.w - bdec[d0 + 3]);
  *(bf16x4*)(Xb + (size_t)g4 * 4) = o;
}

// ---------------- 2) convert W_enc -> bf16 ----------------
__global__ void k_convert_w(const float* __restrict__ w, bf16_t* __restrict__ Wb) {
  int g4 = blockIdx.x * blockDim.x + threadIdx.x;
  float4 v = ((const float4*)w)[g4];
  bf16x4 o;
  o[0] = (bf16_t)v.x; o[1] = (bf16_t)v.y; o[2] = (bf16_t)v.z; o[3] = (bf16_t)v.w;
  *(bf16x4*)(Wb + (size_t)g4 * 4) = o;
}

// ---------------- 3) WMMA GEMM: acts = relu(Xb @ Wb^T + b_enc) ----------------
// block tile 128(M) x 128(N), K-step 32. 256 threads = 8 waves,
// wave grid 4(M-dir, 32 rows each) x 2(N-dir, 64 cols each) -> 2x4 sixteen-tiles/wave.
#define LDT 40   // LDS row stride in halfs (80B) -> conflict-free b128 frag reads

__global__ __launch_bounds__(256) void k_gemm(const bf16_t* __restrict__ Xb,
                                              const bf16_t* __restrict__ Wb,
                                              const float*  __restrict__ b_enc,
                                              float* __restrict__ acts) {
  const int tid  = threadIdx.x;
  const int lane = tid & 31;
  const int wave = tid >> 5;
  const int m0 = blockIdx.y * 128;
  const int n0 = blockIdx.x * 128;
  const int wm = (wave >> 1) * 32;   // 0,32,64,96
  const int wn = (wave & 1) * 64;    // 0,64
  const int trow = tid >> 1;
  const int tcol = (tid & 1) * 16;   // halfs

  v8f acc[2][4];
#pragma unroll
  for (int i = 0; i < 2; ++i)
#pragma unroll
    for (int j = 0; j < 4; ++j)
#pragma unroll
      for (int e = 0; e < 8; ++e) acc[i][j][e] = 0.0f;

  const int kb  = (lane >> 4) * 8;          // K sub-chunk base (halfs)
  const int rA0 = wm + (lane & 15);
  const int rB0 = wn + (lane & 15);

  union V16 { bf16x8 h[2]; v16bf v; };

  const bf16_t* gA_base = Xb + (size_t)(m0 + trow) * D_IN + tcol;
  const bf16_t* gB_base = Wb + (size_t)(n0 + trow) * D_IN + tcol;

#if HAVE_ASYNC_LDS
  // ---- double-buffered async-to-LDS pipeline (ASYNCcnt tracked DMA) ----
  __shared__ bf16_t As[2][128 * LDT];
  __shared__ bf16_t Bs[2][128 * LDT];

  auto issue = [&](int kt, int buf) {
    const bf16_t* ga = gA_base + kt * 32;
    const bf16_t* gb = gB_base + kt * 32;
    bf16_t* la = &As[buf][trow * LDT + tcol];
    bf16_t* lb = &Bs[buf][trow * LDT + tcol];
    async_b128(ga,     la);
    async_b128(ga + 8, la + 8);
    async_b128(gb,     lb);
    async_b128(gb + 8, lb + 8);
  };

  issue(0, 0);
  issue(1, 1);

  for (int kt = 0; kt < D_IN / 32; ++kt) {
    const int cur = kt & 1;
    // in-order completion: cnt<=4 leaves only the next buffer's 4 loads pending
    if (kt < D_IN / 32 - 1) { WAIT_ASYNC(4); } else { WAIT_ASYNC(0); }
    __syncthreads();                       // current buffer visible to all waves

    V16 afr[2], bfr[4];
#pragma unroll
    for (int i = 0; i < 2; ++i) {
      int r = rA0 + i * 16;
      afr[i].h[0] = *(const bf16x8*)&As[cur][r * LDT + kb];
      afr[i].h[1] = *(const bf16x8*)&As[cur][r * LDT + kb + 16];
    }
#pragma unroll
    for (int j = 0; j < 4; ++j) {
      int r = rB0 + j * 16;
      bfr[j].h[0] = *(const bf16x8*)&Bs[cur][r * LDT + kb];
      bfr[j].h[1] = *(const bf16x8*)&Bs[cur][r * LDT + kb + 16];
    }
#pragma unroll
    for (int i = 0; i < 2; ++i)
#pragma unroll
      for (int j = 0; j < 4; ++j)
        acc[i][j] = __builtin_amdgcn_wmma_f32_16x16x32_bf16(
            false, afr[i].v, false, bfr[j].v, (short)0, acc[i][j], false, false);

    __syncthreads();                       // all ds reads of `cur` drained to VGPRs
    if (kt + 2 < D_IN / 32) issue(kt + 2, cur);   // overwrite freed buffer
  }
#else
  // ---- fallback: single-buffer staging through VGPRs ----
  __shared__ bf16_t As[128 * LDT];
  __shared__ bf16_t Bs[128 * LDT];

  for (int kt = 0; kt < D_IN / 32; ++kt) {
    const int kg = kt * 32;
    const bf16x8* gA = (const bf16x8*)(gA_base + kg);
    const bf16x8* gB = (const bf16x8*)(gB_base + kg);
    bf16x8 a0 = gA[0], a1 = gA[1];
    bf16x8 b0 = gB[0], b1 = gB[1];
    *(bf16x8*)&As[trow * LDT + tcol]     = a0;
    *(bf16x8*)&As[trow * LDT + tcol + 8] = a1;
    *(bf16x8*)&Bs[trow * LDT + tcol]     = b0;
    *(bf16x8*)&Bs[trow * LDT + tcol + 8] = b1;
    __syncthreads();

    V16 afr[2], bfr[4];
#pragma unroll
    for (int i = 0; i < 2; ++i) {
      int r = rA0 + i * 16;
      afr[i].h[0] = *(const bf16x8*)&As[r * LDT + kb];
      afr[i].h[1] = *(const bf16x8*)&As[r * LDT + kb + 16];
    }
#pragma unroll
    for (int j = 0; j < 4; ++j) {
      int r = rB0 + j * 16;
      bfr[j].h[0] = *(const bf16x8*)&Bs[r * LDT + kb];
      bfr[j].h[1] = *(const bf16x8*)&Bs[r * LDT + kb + 16];
    }
#pragma unroll
    for (int i = 0; i < 2; ++i)
#pragma unroll
      for (int j = 0; j < 4; ++j)
        acc[i][j] = __builtin_amdgcn_wmma_f32_16x16x32_bf16(
            false, afr[i].v, false, bfr[j].v, (short)0, acc[i][j], false, false);
    __syncthreads();
  }
#endif

  // epilogue: + b_enc, ReLU, store f32
#pragma unroll
  for (int j = 0; j < 4; ++j) {
    const int nj = n0 + wn + j * 16 + (lane & 15);
    const float be = b_enc[nj];
#pragma unroll
    for (int i = 0; i < 2; ++i) {
      const int mb = m0 + wm + i * 16 + ((lane >> 4) << 3);
#pragma unroll
      for (int v = 0; v < 8; ++v) {
        float val = acc[i][j][v] + be;
        val = val > 0.f ? val : 0.f;
        acts[(size_t)(mb + v) * NLAT + nj] = val;
      }
    }
  }
}

// ---------------- 4) column stats of x (for total_variance) ----------------
__global__ void k_colstats(const float* __restrict__ x, float* __restrict__ colsum,
                           float* __restrict__ colsumsq) {
  int d = blockIdx.x * blockDim.x + threadIdx.x;   // 0..1023
  float s = 0.f, s2 = 0.f;
  for (int n = 0; n < N_TOKENS; ++n) {
    float v = x[(size_t)n * D_IN + d];
    s += v; s2 += v * v;
  }
  colsum[d] = s; colsumsq[d] = s2;
}

// ---------------- 5) per-token radix top-64 + sparse decode ----------------
__global__ __launch_bounds__(256) void k_topk_decode(
    const float* __restrict__ acts, const float* __restrict__ x,
    const float* __restrict__ Wdec, const float* __restrict__ bdec,
    float* __restrict__ sae_out, float* __restrict__ top_v_out,
    float* __restrict__ top_i_out, float* __restrict__ tok_e2) {
  extern __shared__ unsigned char smem[];
  float*    row  = (float*)smem;            // 32768
  unsigned* hist = (unsigned*)(row + NLAT); // 256
  float*    cv   = (float*)(hist + 256);    // 64 candidate vals
  int*      ci   = (int*)(cv + 64);         // 64 candidate idx
  int*      eqi  = ci + 64;                 // 128 tie idx
  float*    sv   = (float*)(eqi + 128);     // 64 sorted vals
  int*      si   = (int*)(sv + 64);         // 64 sorted idx
  float*    red  = (float*)(si + 64);       // 256 reduction
  int*      scal = (int*)(red + 256);       // 4 scalars

  const int tid = threadIdx.x;
  const int n   = blockIdx.x;
  const float* arow = acts + (size_t)n * NLAT;

  // stage full activation row into LDS (128 KB)
  {
    const float4* a4 = (const float4*)arow;
    float4* r4 = (float4*)row;
    for (int i = tid; i < NLAT / 4; i += 256) r4[i] = a4[i];
  }
  __syncthreads();

  // ---- radix select kth-largest (values >= 0 so uint bits are monotone) ----
  unsigned prefix = 0;
  int kneed = TOPK;
  for (int byte = 3; byte >= 0; --byte) {
    hist[tid] = 0;
    __syncthreads();
    const int shift = byte * 8;
    for (int i = tid; i < NLAT; i += 256) {
      unsigned v = __float_as_uint(row[i]);
      bool match = (byte == 3) || ((v >> (shift + 8)) == prefix);
      if (match) atomicAdd(&hist[(v >> shift) & 0xFF], 1u);
    }
    __syncthreads();
    if (tid == 0) {
      int cum = 0, d;
      for (d = 255; d >= 0; --d) {
        int c = (int)hist[d];
        if (cum + c >= kneed) break;
        cum += c;
      }
      scal[0] = d; scal[1] = cum;
    }
    __syncthreads();
    prefix = (prefix << 8) | (unsigned)scal[0];
    kneed -= scal[1];
  }
  const unsigned Tbits = prefix;   // bits of kth-largest value; kneed = #ties to take

  // ---- collect: all strictly greater, plus kneed smallest-index ties ----
  if (tid == 0) { scal[0] = 0; scal[1] = 0; }
  __syncthreads();
  for (int i = tid; i < NLAT; i += 256) {
    unsigned v = __float_as_uint(row[i]);
    if (v > Tbits) {
      int p = atomicAdd(&scal[0], 1);
      if (p < TOPK) { cv[p] = row[i]; ci[p] = i; }
    } else if (v == Tbits) {
      int p = atomicAdd(&scal[1], 1);
      if (p < 128) eqi[p] = i;
    }
  }
  __syncthreads();
  const int ngt = min(scal[0], TOPK);
  const int neq = min(scal[1], 128);
  if (tid < neq) {
    int mi = eqi[tid];
    int rank = 0;
    for (int j = 0; j < neq; ++j) rank += (eqi[j] < mi) ? 1 : 0;
    if (rank < kneed && (ngt + rank) < TOPK) {
      cv[ngt + rank] = __uint_as_float(Tbits);
      ci[ngt + rank] = mi;
    }
  }
  __syncthreads();

  // ---- rank sort 64 (descending value, ascending index — jax tie order) ----
  if (tid < TOPK) {
    unsigned mv = __float_as_uint(cv[tid]);
    int mi = ci[tid];
    int r = 0;
    for (int j = 0; j < TOPK; ++j) {
      unsigned jv = __float_as_uint(cv[j]);
      int ji = ci[j];
      r += ((jv > mv) || (jv == mv && ji < mi)) ? 1 : 0;
    }
    sv[r] = cv[tid]; si[r] = ci[tid];
  }
  __syncthreads();

  if (tid < TOPK) {
    top_v_out[(size_t)n * TOPK + tid] = sv[tid];
    top_i_out[(size_t)n * TOPK + tid] = (float)si[tid];
  }

  // ---- sparse decode + per-token squared error ----
  float e2 = 0.f;
  for (int d = tid; d < D_IN; d += 256) {
    float a = bdec[d];
#pragma unroll 4
    for (int j = 0; j < TOPK; ++j)
      a += sv[j] * Wdec[(size_t)si[j] * D_IN + d];
    sae_out[(size_t)n * D_IN + d] = a;
    float e = a - x[(size_t)n * D_IN + d];
    e2 += e * e;
  }
  red[tid] = e2;
  __syncthreads();
  for (int s = 128; s > 0; s >>= 1) {
    if (tid < s) red[tid] += red[tid + s];
    __syncthreads();
  }
  if (tid == 0) tok_e2[n] = red[0];
}

// ---------------- 6) fvu scalar ----------------
__global__ void k_fvu(const float* __restrict__ colsum, const float* __restrict__ colsumsq,
                      const float* __restrict__ tok_e2, float* __restrict__ out_fvu) {
  __shared__ float rtv[256], rse[256];
  int tid = threadIdx.x;
  float tv = 0.f;
  for (int d = tid; d < D_IN; d += 256) {
    float s = colsum[d];
    tv += colsumsq[d] - s * s * (1.0f / (float)N_TOKENS);
  }
  float se = 0.f;
  for (int i = tid; i < N_TOKENS; i += 256) se += tok_e2[i];
  rtv[tid] = tv; rse[tid] = se;
  __syncthreads();
  for (int s = 128; s > 0; s >>= 1) {
    if (tid < s) { rtv[tid] += rtv[tid + s]; rse[tid] += rse[tid + s]; }
    __syncthreads();
  }
  if (tid == 0) out_fvu[0] = rse[0] / rtv[0];
}

// ---------------- launch ----------------
extern "C" void kernel_launch(void* const* d_in, const int* in_sizes, int n_in,
                              void* d_out, int out_size, void* d_ws, size_t ws_size,
                              hipStream_t stream) {
  const float* x     = (const float*)d_in[0];
  const float* W_enc = (const float*)d_in[1];
  const float* b_enc = (const float*)d_in[2];
  const float* W_dec = (const float*)d_in[3];
  const float* b_dec = (const float*)d_in[4];
  (void)in_sizes; (void)n_in; (void)out_size; (void)ws_size;   // k fixed at 64

  char* ws = (char*)d_ws;
  float*  acts    = (float*)ws;
  bf16_t* Xb      = (bf16_t*)(ws + XB_OFF);
  bf16_t* Wb      = (bf16_t*)(ws + WB_OFF);
  float*  colsum  = (float*)(ws + CS_OFF);
  float*  colsumq = (float*)(ws + CQ_OFF);
  float*  tok_e2  = (float*)(ws + E2_OFF);

  float* out     = (float*)d_out;
  float* sae_out = out + OUT_SAE;
  float* top_v   = out + OUT_TV;
  float* top_i   = out + OUT_TI;
  float* fvu     = out + OUT_FVU;

  // 1) bf16 conversions
  k_convert_x<<<(N_TOKENS * D_IN) / (256 * 4), 256, 0, stream>>>(x, b_dec, Xb);
  k_convert_w<<<(NLAT * D_IN) / (256 * 4), 256, 0, stream>>>(W_enc, Wb);

  // 2) encoder GEMM (WMMA bf16 -> f32), fused bias+ReLU
  dim3 ggrid(NLAT / 128, N_TOKENS / 128);
  k_gemm<<<ggrid, 256, 0, stream>>>(Xb, Wb, b_enc, acts);

  // 3) column stats for total_variance
  k_colstats<<<D_IN / 256, 256, 0, stream>>>(x, colsum, colsumq);

  // 4) per-token top-k + decode
  size_t smem = (size_t)NLAT * 4       // row
              + 256 * 4                // hist
              + 64 * 4 + 64 * 4        // cv, ci
              + 128 * 4                // eqi
              + 64 * 4 + 64 * 4        // sv, si
              + 256 * 4                // red
              + 16;                    // scalars
  k_topk_decode<<<N_TOKENS, 256, smem, stream>>>(acts, x, W_dec, b_dec,
                                                 sae_out, top_v, top_i, tok_e2);

  // 5) fvu
  k_fvu<<<1, 256, 0, stream>>>(colsum, colsumq, tok_e2, fvu);
}
